// TernaryLinear_65661460021995
// MI455X (gfx1250) — matrix operand order repaired and checked by reading, not compiled, not run on hardware
//
#include <hip/hip_runtime.h>
#include <hip/hip_bf16.h>

typedef __attribute__((ext_vector_type(16))) _Float16 v16h;
typedef __attribute__((ext_vector_type(8)))  float    v8f;

#define BATCH   524288
#define NFEAT   128
#define EPSBN   1e-5f

// ---------------------------------------------------------------------------
// Kernel 1: per-column partial sums / sums-of-squares.
// 256 blocks x 256 threads; block handles 2048 rows; thread (col = tid&127,
// row parity = tid>>7) accumulates 1024 rows. Reads are fully coalesced
// (consecutive threads -> consecutive columns). Deterministic (no atomics).
// ---------------------------------------------------------------------------
__global__ void __launch_bounds__(256)
ternlin_stats_kernel(const float* __restrict__ x,
                     float* __restrict__ psum, float* __restrict__ psumsq) {
    const int tid  = threadIdx.x;
    const int col  = tid & 127;
    const int half = tid >> 7;                       // 0 or 1
    const size_t rowBase = (size_t)blockIdx.x * 2048 + half;
    const float* xp = x + rowBase * NFEAT + col;
    float s = 0.0f, q = 0.0f;
    for (int it = 0; it < 1024; ++it) {              // stride 2 rows
        float v = xp[(size_t)it * (2 * NFEAT)];
        s += v;
        q = fmaf(v, v, q);
    }
    __shared__ float ls[256], lq[256];
    ls[tid] = s; lq[tid] = q;
    __syncthreads();
    if (tid < 128) {
        psum  [blockIdx.x * 128 + tid] = ls[tid] + ls[tid + 128];
        psumsq[blockIdx.x * 128 + tid] = lq[tid] + lq[tid + 128];
    }
}

// ---------------------------------------------------------------------------
// Kernel 2: finalize stats, fold BN into weights/bias, and pre-swizzle the
// f16 weight matrix into WMMA B-fragment order:
//   fragTab[(kc*8+n)*32 + lane][v] = pack_f16( w2[o][k0], w2[o][k0+1] )
//   o  = n*16 + (lane&15)
//   k0 = kc*32 + (lane>>4)*8 + 2*v + (v>>2)*8      (ISA 7.12.2 16-bit B layout)
// ---------------------------------------------------------------------------
__global__ void __launch_bounds__(256)
ternlin_prep_kernel(const float* __restrict__ psum, const float* __restrict__ psumsq,
                    const float* __restrict__ weight, const float* __restrict__ bias,
                    const float* __restrict__ gamma,  const float* __restrict__ beta,
                    const float* __restrict__ alpha,
                    unsigned int* __restrict__ fragTab, float* __restrict__ bias2) {
    __shared__ float s_s[128];   // per-input scale  = gamma * rsqrt(var+eps)
    __shared__ float s_t[128];   // per-input offset = beta - mean*scale
    const int tid = threadIdx.x;

    if (tid < 128) {
        float s = 0.0f, q = 0.0f;
        for (int b = 0; b < 256; ++b) {
            s += psum  [b * 128 + tid];
            q += psumsq[b * 128 + tid];
        }
        const float invB = 1.0f / (float)BATCH;
        float mean = s * invB;
        float var  = fmaxf(q * invB - mean * mean, 0.0f);   // biased variance
        float sc   = gamma[tid] * rsqrtf(var + EPSBN);
        s_s[tid] = sc;
        s_t[tid] = beta[tid] - mean * sc;
    }
    __syncthreads();

    // bias2[o] = bias[o] + sum_k t[k] * sign(weight[o][k] - alpha[k])
    if (tid < 128) {
        float acc = bias[tid];
        for (int k = 0; k < 128; ++k) {
            float w = (weight[tid * 128 + k] - alpha[k]) > 0.0f ? 1.0f : -1.0f;
            acc = fmaf(s_t[k], w, acc);
        }
        bias2[tid] = acc;
    }

    // 8192 packed-u32 fragment entries (4 kc * 8 n * 32 lanes * 8 vgprs)
    for (int e = tid; e < 8192; e += 256) {
        const int kc = e >> 11;
        const int n  = (e >> 8) & 7;
        const int l  = (e >> 3) & 31;
        const int v  = e & 7;
        const int o  = n * 16 + (l & 15);
        const int k0 = kc * 32 + ((l >> 4) << 3) + 2 * v + ((v >> 2) << 3);
        float w0 = ((weight[o * 128 + k0    ] - alpha[k0    ]) > 0.0f ? 1.0f : -1.0f) * s_s[k0    ];
        float w1 = ((weight[o * 128 + k0 + 1] - alpha[k0 + 1]) > 0.0f ? 1.0f : -1.0f) * s_s[k0 + 1];
        union { _Float16 h[2]; unsigned int u; } p;
        p.h[0] = (_Float16)w0;   // low  half  <-> even K (VGPR element 2v)
        p.h[1] = (_Float16)w1;   // high half  <-> odd  K (VGPR element 2v+1)
        fragTab[e] = p.u;
    }
}

// ---------------------------------------------------------------------------
// Kernel 3: GEMM  out[b][o] = sum_i x[b][i]*w2[o][i] + bias2[o]
// 8 waves / block; each wave owns one 16-row M tile and all 128 N columns.
// Weight fragments staged once per block into LDS (32 KB), then consumed with
// ds_load_b128. 4 K-chunks x 8 N-tiles = 32 v_wmma_f32_16x16x32_f16 per tile.
// ---------------------------------------------------------------------------
__global__ void __launch_bounds__(256)
ternlin_gemm_kernel(const float* __restrict__ x,
                    const unsigned int* __restrict__ fragTab,
                    const float* __restrict__ bias2,
                    float* __restrict__ out) {
    __shared__ unsigned int lfrag[8192];             // 32 KB (of 320 KB/WGP)
    const int tid = threadIdx.x;

    {   // cooperative 32 KB stage-in, 16B per load
        const uint4* src = (const uint4*)fragTab;
        uint4*       dst = (uint4*)lfrag;
        for (int i = tid; i < 2048; i += 256) dst[i] = src[i];
    }
    __syncthreads();

    const int wave = tid >> 5;
    const int lane = tid & 31;
    const int tile = blockIdx.x * 8 + wave;          // 4096 blocks * 8 = 32768 tiles
    const int row  = tile * 16 + (lane & 15);        // A-matrix row for this lane
    const int hsel = (lane >> 4);                    // 0: K base +0, 1: K base +8

    v8f acc[8] = {};                                 // 16x128 f32 accumulators

    const float* xrow = x + (size_t)row * NFEAT;
#pragma unroll
    for (int kc = 0; kc < 4; ++kc) {
        // A fragment, ISA 16-bit A layout:
        //   elements 0..7  <-> K = kc*32 + hsel*8 + (0..7)
        //   elements 8..15 <-> K = kc*32 + hsel*8 + 16 + (0..7)
        const float4* xa = (const float4*)(xrow + kc * 32 + hsel * 8);
        float4 f0 = xa[0], f1 = xa[1];               // K +0..7
        float4 f2 = xa[4], f3 = xa[5];               // K +16..23
        v16h a;
        a[0]  = (_Float16)f0.x; a[1]  = (_Float16)f0.y;
        a[2]  = (_Float16)f0.z; a[3]  = (_Float16)f0.w;
        a[4]  = (_Float16)f1.x; a[5]  = (_Float16)f1.y;
        a[6]  = (_Float16)f1.z; a[7]  = (_Float16)f1.w;
        a[8]  = (_Float16)f2.x; a[9]  = (_Float16)f2.y;
        a[10] = (_Float16)f2.z; a[11] = (_Float16)f2.w;
        a[12] = (_Float16)f3.x; a[13] = (_Float16)f3.y;
        a[14] = (_Float16)f3.z; a[15] = (_Float16)f3.w;

#pragma unroll
        for (int n = 0; n < 8; ++n) {
            const uint4* bp = (const uint4*)&lfrag[((kc * 8 + n) * 32 + lane) * 8];
            union { uint4 q[2]; v16h h; } bb;
            bb.q[0] = bp[0];
            bb.q[1] = bp[1];
            acc[n] = __builtin_amdgcn_wmma_f32_16x16x32_f16(
                /*neg_a=*/false, a, /*neg_b=*/false, bb.h,
                /*c_mod=*/(short)0, acc[n], /*reuse_a=*/false, /*reuse_b=*/false);
        }
    }

    // Epilogue: C/D layout — lane holds N = lane&15, VGPR j -> M = j (+8 for
    // upper half-wave). Half-wave writes 64B contiguous per (n, j).
    const int col0  = lane & 15;
    const int rbase = tile * 16 + (hsel << 3);
#pragma unroll
    for (int n = 0; n < 8; ++n) {
        const float bv = bias2[n * 16 + col0];
#pragma unroll
        for (int j = 0; j < 8; ++j) {
            out[(size_t)(rbase + j) * NFEAT + n * 16 + col0] = acc[n][j] + bv;
        }
    }
}

// ---------------------------------------------------------------------------
// Launch. Inputs: x, weight, bias, gamma, beta, alpha (all f32).
// ws layout (floats): psum[256*128] | psumsq[256*128] | fragTab[8192 u32] |
// bias2[128]  -> ~295 KB total.
// ---------------------------------------------------------------------------
extern "C" void kernel_launch(void* const* d_in, const int* in_sizes, int n_in,
                              void* d_out, int out_size, void* d_ws, size_t ws_size,
                              hipStream_t stream) {
    const float* x      = (const float*)d_in[0];
    const float* weight = (const float*)d_in[1];
    const float* bias   = (const float*)d_in[2];
    const float* gamma  = (const float*)d_in[3];
    const float* beta   = (const float*)d_in[4];
    const float* alpha  = (const float*)d_in[5];
    float* out = (float*)d_out;

    float*        ws      = (float*)d_ws;
    float*        psum    = ws;                       // 32768 floats
    float*        psumsq  = ws + 256 * 128;           // 32768 floats
    unsigned int* fragTab = (unsigned int*)(ws + 2 * 256 * 128); // 8192 u32
    float*        bias2   = ws + 2 * 256 * 128 + 8192;           // 128 floats

    ternlin_stats_kernel<<<256, 256, 0, stream>>>(x, psum, psumsq);
    ternlin_prep_kernel<<<1, 256, 0, stream>>>(psum, psumsq, weight, bias,
                                               gamma, beta, alpha, fragTab, bias2);
    ternlin_gemm_kernel<<<4096, 256, 0, stream>>>(x, fragTab, bias2, out);
}
